// LearnableGlobalLocalMultiheadAttention_71863392796957
// MI455X (gfx1250) — matrix operand order repaired and checked by reading, not compiled, not run on hardware
//
#include <hip/hip_runtime.h>

typedef _Float16 v16h __attribute__((ext_vector_type(16)));
typedef float    v8f  __attribute__((ext_vector_type(8)));
typedef unsigned int u32x4 __attribute__((ext_vector_type(4)));
typedef int          i32x4 __attribute__((ext_vector_type(4)));
typedef int          i32x8 __attribute__((ext_vector_type(8)));

#define EMB 1024
#define NH  16
#define HD  64
#define LL  1024
#define MM  4096          // B*L
#define SCALING 0.125f
#define HEADSZ 65536      // L*HD per (b,h)
#define ARR4M  4194304    // BH*L*HD == B*L*E

// A-fragment (16x32 f16) element->K mapping: contiguous, k = 16*h + e (matches B fills)
__device__ __forceinline__ int amap(int e, int h) {
    return h * 16 + e;
}

__device__ __forceinline__ v8f wmma16(v16h a, v16h b, v8f c) {
    return __builtin_amdgcn_wmma_f32_16x16x32_f16(false, a, false, b, (short)0, c,
                                                  false, false);
}

__device__ __forceinline__ void zero8(v8f& x) {
#pragma unroll
    for (int i = 0; i < 8; i++) x[i] = 0.0f;
}

// Issue one TDM DMA: 2D f16 tile (tile_d1 rows x tile_d0 elems, row stride
// d0_stride elems) from global memory -> LDS at byte offset lds_off.
// D# packing per CDNA5 ISA 8.3/8.4 (group0: count/lds/global/type,
// group1: data_size=2B, dims/strides). Groups 2/3 zero (2D tensor).
__device__ __forceinline__ void tdm_load_2d_f16(const _Float16* gptr,
                                                unsigned lds_off,
                                                unsigned tensor_d0,
                                                unsigned tensor_d1,
                                                unsigned d0_stride,
                                                unsigned tile_d0,
                                                unsigned tile_d1) {
    unsigned long long ga = (unsigned long long)(size_t)gptr;
    u32x4 g0;
    g0[0] = 1u;                                    // count=1, user mode
    g0[1] = lds_off;                               // lds_addr (bytes)
    g0[2] = (unsigned)(ga & 0xffffffffull);        // global_addr[31:0]
    g0[3] = (unsigned)((ga >> 32) & 0x01ffffffull) | (2u << 30);  // type=2
    i32x8 g1;
    g1[0] = (int)(1u << 16);                       // data_size=1 -> 2 bytes
    g1[1] = (int)((tensor_d0 & 0xffffu) << 16);    // tensor_dim0[15:0]
    g1[2] = (int)(((tensor_d0 >> 16) & 0xffffu) | ((tensor_d1 & 0xffffu) << 16));
    g1[3] = (int)(((tensor_d1 >> 16) & 0xffffu) | ((tile_d0 & 0xffffu) << 16));
    g1[4] = (int)(tile_d1 & 0xffffu);              // tile_dim1, tile_dim2=0
    g1[5] = (int)d0_stride;                        // tensor_dim0_stride[31:0]
    g1[6] = 0;
    g1[7] = 0;
    i32x4 z4;
    z4[0] = 0; z4[1] = 0; z4[2] = 0; z4[3] = 0;
    i32x8 z8;
#pragma unroll
    for (int i = 0; i < 8; i++) z8[i] = 0;
    __builtin_amdgcn_tensor_load_to_lds(g0, g1, z4, z4, z8, 0);
}

// -------------------------------------------------------------------------
// Zero-init for the atomic-accumulated consistent_mask region of d_out.
__global__ __launch_bounds__(256) void zero_kernel(float* __restrict__ p, int n) {
    int i = blockIdx.x * 256 + threadIdx.x;
    if (i < n) p[i] = 0.0f;
}

// -------------------------------------------------------------------------
// Projections: Y[slot] = X_slot @ W[slot*E:(slot+1)*E]^T + b, slot0 scaled.
// Output stored split-head f16: ws[slot][(b*16+h)][l][d].
// Block: 128 thr (4 waves), tile 64(M)x64(N), K staged 32-wide in LDS.
__global__ __launch_bounds__(128) void proj_kernel(
    const float* __restrict__ query, const float* __restrict__ key_in,
    const float* __restrict__ value, const float* __restrict__ W,
    const float* __restrict__ bias, _Float16* __restrict__ ws) {
    __shared__ _Float16 Xs[64][34];
    __shared__ _Float16 Wl[64][34];
    const int slot = blockIdx.z;
    const float* src = (slot == 2) ? value : ((slot == 1 || slot == 4) ? key_in : query);
    const float scale = (slot == 0) ? SCALING : 1.0f;
    const int m0 = blockIdx.x * 64, n0 = blockIdx.y * 64;
    const int t = threadIdx.x, w = t >> 5, lane = t & 31;
    const int half = lane >> 4, ln = lane & 15;
    const int lrow = t >> 1, lcol = (t & 1) * 16;

    v8f acc[4];
#pragma unroll
    for (int i = 0; i < 4; i++) zero8(acc[i]);

    for (int kb = 0; kb < EMB; kb += 32) {
        __builtin_prefetch(src + (m0 + lrow) * EMB + kb + 32, 0, 1);
#pragma unroll
        for (int i = 0; i < 16; i++) {
            Xs[lrow][lcol + i] = (_Float16)src[(m0 + lrow) * EMB + kb + lcol + i];
            Wl[lrow][lcol + i] =
                (_Float16)W[(slot * EMB + n0 + lrow) * EMB + kb + lcol + i];
        }
        __syncthreads();
        v16h a;
#pragma unroll
        for (int e = 0; e < 16; e++) a[e] = Xs[w * 16 + ln][amap(e, half)];
#pragma unroll
        for (int nc = 0; nc < 4; nc++) {
            v16h bfr;
#pragma unroll
            for (int e = 0; e < 16; e++) bfr[e] = Wl[nc * 16 + ln][half * 16 + e];
            acc[nc] = wmma16(a, bfr, acc[nc]);
        }
        __syncthreads();
    }

    _Float16* dst = ws + slot * ARR4M;
#pragma unroll
    for (int nc = 0; nc < 4; nc++) {
        int n = n0 + nc * 16 + ln;
        float bv = bias[slot * EMB + n];
#pragma unroll
        for (int vv = 0; vv < 8; vv++) {
            int m = m0 + w * 16 + vv + 8 * half;
            int b = m >> 10, l = m & 1023, h = n >> 6, d = n & 63;
            dst[(((b * NH + h) * LL + l) << 6) + d] =
                (_Float16)((acc[nc][vv] + bv) * scale);
        }
    }
}

// -------------------------------------------------------------------------
// Causal flash attention. One wave per (bh, 16-row q tile); key tiles of 32.
// K/V tiles staged into per-wave LDS buffers via the Tensor Data Mover.
__global__ __launch_bounds__(128) void attn_kernel(
    const _Float16* __restrict__ q, const _Float16* __restrict__ k,
    const _Float16* __restrict__ v, _Float16* __restrict__ ctx) {
    __shared__ _Float16 Kl[4][32][64];   // 16 KB
    __shared__ _Float16 Vl[4][32][64];   // 16 KB
    __shared__ _Float16 plds[4][16][34]; // 4.3 KB
    const int w = threadIdx.x >> 5, lane = threadIdx.x & 31;
    const int half = lane >> 4, ln = lane & 15;
    const int gw = blockIdx.x * 4 + w;
    const int bh = gw >> 6, qt = gw & 63;
    const _Float16* qb = q + bh * HEADSZ;
    const _Float16* kp = k + bh * HEADSZ;
    const _Float16* vp = v + bh * HEADSZ;
    const unsigned koff = (unsigned)(unsigned long long)(size_t)&Kl[w][0][0];
    const unsigned voff = (unsigned)(unsigned long long)(size_t)&Vl[w][0][0];

    v16h a0, a1;
    const int qrow = qt * 16 + ln;
#pragma unroll
    for (int e = 0; e < 16; e++) {
        int kk = amap(e, half);
        a0[e] = qb[qrow * HD + kk];
        a1[e] = qb[qrow * HD + 32 + kk];
    }

    v8f oacc[4];
#pragma unroll
    for (int i = 0; i < 4; i++) zero8(oacc[i]);
    float rowmax[8], rowsum[8];
#pragma unroll
    for (int i = 0; i < 8; i++) { rowmax[i] = -INFINITY; rowsum[i] = 0.0f; }

    const int ktmax = (qt * 16 + 15) >> 5;
    for (int kt = 0; kt <= ktmax; kt++) {
        const int key0 = kt * 32;

        // TDM: stage K and V 32x64 f16 tiles into this wave's LDS buffers
        tdm_load_2d_f16(kp + key0 * HD, koff, HD, 32, HD, HD, 32);
        tdm_load_2d_f16(vp + key0 * HD, voff, HD, 32, HD, HD, 32);
        __builtin_amdgcn_s_wait_tensorcnt(0);
        __builtin_amdgcn_wave_barrier();

        // B fragments of K^T: lane = key column, element = d
        v16h b00, b01, b10, b11;
#pragma unroll
        for (int e = 0; e < 16; e++) {
            int dlo = half * 16 + e;
            b00[e] = Kl[w][ln][dlo];
            b01[e] = Kl[w][ln][32 + dlo];
            b10[e] = Kl[w][16 + ln][dlo];
            b11[e] = Kl[w][16 + ln][32 + dlo];
        }
        v8f s0, s1;
        zero8(s0); zero8(s1);
        s0 = wmma16(a0, b00, s0); s0 = wmma16(a1, b01, s0);
        s1 = wmma16(a0, b10, s1); s1 = wmma16(a1, b11, s1);

        // causal mask
#pragma unroll
        for (int vv = 0; vv < 8; vv++) {
            int mrow = qt * 16 + vv + 8 * half;
            if (key0 + ln > mrow) s0[vv] = -INFINITY;
            if (key0 + 16 + ln > mrow) s1[vv] = -INFINITY;
        }

        // online softmax (rows live in 16-lane halves -> shfl_xor 8/4/2/1)
        float sc8[8];
#pragma unroll
        for (int vv = 0; vv < 8; vv++) {
            float tmx = fmaxf(s0[vv], s1[vv]);
            tmx = fmaxf(tmx, __shfl_xor(tmx, 8, 32));
            tmx = fmaxf(tmx, __shfl_xor(tmx, 4, 32));
            tmx = fmaxf(tmx, __shfl_xor(tmx, 2, 32));
            tmx = fmaxf(tmx, __shfl_xor(tmx, 1, 32));
            float mnew = fmaxf(rowmax[vv], tmx);
            float sc = __expf(rowmax[vv] - mnew);
            float p0 = __expf(s0[vv] - mnew);
            float p1 = __expf(s1[vv] - mnew);
            s0[vv] = p0; s1[vv] = p1;
            float rs = p0 + p1;
            rs += __shfl_xor(rs, 8, 32);
            rs += __shfl_xor(rs, 4, 32);
            rs += __shfl_xor(rs, 2, 32);
            rs += __shfl_xor(rs, 1, 32);
            rowsum[vv] = rowsum[vv] * sc + rs;
            rowmax[vv] = mnew;
            sc8[vv] = sc;
        }
#pragma unroll
        for (int nc = 0; nc < 4; nc++)
#pragma unroll
            for (int vv = 0; vv < 8; vv++) oacc[nc][vv] *= sc8[vv];

        // P (C-layout) -> LDS -> A-fragment layout (f16)
#pragma unroll
        for (int vv = 0; vv < 8; vv++) {
            int r = vv + 8 * half;
            plds[w][r][ln] = (_Float16)s0[vv];
            plds[w][r][16 + ln] = (_Float16)s1[vv];
        }
        __builtin_amdgcn_wave_barrier();
        v16h pa;
#pragma unroll
        for (int e = 0; e < 16; e++) pa[e] = plds[w][ln][amap(e, half)];
        __builtin_amdgcn_wave_barrier();

        // O += P @ V (V B-fragments from LDS tile)
#pragma unroll
        for (int nc = 0; nc < 4; nc++) {
            v16h vf;
#pragma unroll
            for (int e = 0; e < 16; e++) {
                int kk = half * 16 + e;
                vf[e] = Vl[w][kk][nc * 16 + ln];
            }
            oacc[nc] = wmma16(pa, vf, oacc[nc]);
        }
    }

    const int b = bh >> 4, h = bh & 15;
#pragma unroll
    for (int vv = 0; vv < 8; vv++) {
        float inv = 1.0f / rowsum[vv];
        int l = qt * 16 + vv + 8 * half;
#pragma unroll
        for (int nc = 0; nc < 4; nc++) {
            int d = nc * 16 + ln;
            ctx[(b * LL + l) * EMB + h * HD + d] = (_Float16)(oacc[nc][vv] * inv);
        }
    }
}

// -------------------------------------------------------------------------
// Output projection: d_out[:B*L*E] = ctx @ out_w^T + out_b  (fp32 store)
__global__ __launch_bounds__(128) void outproj_kernel(
    const _Float16* __restrict__ ctx, const float* __restrict__ Wo,
    const float* __restrict__ bo, float* __restrict__ out) {
    __shared__ _Float16 Xs[64][34];
    __shared__ _Float16 Wl[64][34];
    const int m0 = blockIdx.x * 64, n0 = blockIdx.y * 64;
    const int t = threadIdx.x, w = t >> 5, lane = t & 31;
    const int half = lane >> 4, ln = lane & 15;
    const int lrow = t >> 1, lcol = (t & 1) * 16;

    v8f acc[4];
#pragma unroll
    for (int i = 0; i < 4; i++) zero8(acc[i]);

    for (int kb = 0; kb < EMB; kb += 32) {
        __builtin_prefetch(Wo + (n0 + lrow) * EMB + kb + 32, 0, 1);
#pragma unroll
        for (int i = 0; i < 16; i++) {
            Xs[lrow][lcol + i] = ctx[(m0 + lrow) * EMB + kb + lcol + i];
            Wl[lrow][lcol + i] = (_Float16)Wo[(n0 + lrow) * EMB + kb + lcol + i];
        }
        __syncthreads();
        v16h a;
#pragma unroll
        for (int e = 0; e < 16; e++) a[e] = Xs[w * 16 + ln][amap(e, half)];
#pragma unroll
        for (int nc = 0; nc < 4; nc++) {
            v16h bfr;
#pragma unroll
            for (int e = 0; e < 16; e++) bfr[e] = Wl[nc * 16 + ln][half * 16 + e];
            acc[nc] = wmma16(a, bfr, acc[nc]);
        }
        __syncthreads();
    }

#pragma unroll
    for (int nc = 0; nc < 4; nc++) {
        int n = n0 + nc * 16 + ln;
        float bv = bo[n];
#pragma unroll
        for (int vv = 0; vv < 8; vv++) {
            int m = m0 + w * 16 + vv + 8 * half;
            out[m * EMB + n] = acc[nc][vv] + bv;
        }
    }
}

// -------------------------------------------------------------------------
// Local masking: per (bh, 16-row tile) compute att strip via WMMA into LDS,
// argmax + threshold per row, atomically accumulate 0/1 mask over heads.
// Condition simplified: dm*att < mean  <=>  |j-i0|*att < sum(relu*|j-i0|)/(L-1)
__global__ __launch_bounds__(256) void mask_kernel(
    const _Float16* __restrict__ qm, const _Float16* __restrict__ km,
    float* __restrict__ consistent) {
    __shared__ float att[16][1024];   // 64 KB strip
    const int w = threadIdx.x >> 5, lane = threadIdx.x & 31;
    const int half = lane >> 4, ln = lane & 15;
    const int bh = blockIdx.x >> 6, it = blockIdx.x & 63;
    const _Float16* qb = qm + bh * HEADSZ;
    const _Float16* kb = km + bh * HEADSZ;

    v16h a0, a1;
    const int r0 = it * 16 + ln;
#pragma unroll
    for (int e = 0; e < 16; e++) {
        int kk = amap(e, half);
        a0[e] = qb[r0 * HD + kk];
        a1[e] = qb[r0 * HD + 32 + kk];
    }
    for (int jt = w; jt < 64; jt += 8) {
        v16h b0, b1;
#pragma unroll
        for (int e = 0; e < 16; e++) {
            int d = half * 16 + e;
            b0[e] = kb[(jt * 16 + ln) * HD + d];
            b1[e] = kb[(jt * 16 + ln) * HD + 32 + d];
        }
        v8f s;
        zero8(s);
        s = wmma16(a0, b0, s);
        s = wmma16(a1, b1, s);
#pragma unroll
        for (int vv = 0; vv < 8; vv++) att[vv + 8 * half][jt * 16 + ln] = s[vv];
    }
    __syncthreads();

    const int r = threadIdx.x >> 4, part = threadIdx.x & 15;
    const int j0 = part * 64;

    float vmax = -INFINITY; int imax = 0;
    for (int jj = 0; jj < 64; jj++) {
        float a = att[r][j0 + jj];
        if (a > vmax) { vmax = a; imax = j0 + jj; }
    }
    const int masks[4] = {1, 2, 4, 8};
#pragma unroll
    for (int i = 0; i < 4; i++) {
        float ov = __shfl_xor(vmax, masks[i], 32);
        int oi = __shfl_xor(imax, masks[i], 32);
        if (ov > vmax || (ov == vmax && oi < imax)) { vmax = ov; imax = oi; }
    }

    float tsum = 0.0f;
    for (int jj = 0; jj < 64; jj++) {
        int j = j0 + jj;
        float a = att[r][j];
        float ar = a > 0.0f ? a : 0.0f;
        tsum += ar * fabsf((float)(j - imax));
    }
#pragma unroll
    for (int i = 0; i < 4; i++) tsum += __shfl_xor(tsum, masks[i], 32);
    const float T = tsum * (1.0f / 1023.0f);

    float* crow = consistent + (it * 16 + r) * LL;
    for (int jj = 0; jj < 64; jj++) {
        int j = j0 + jj;
        float a = att[r][j];
        if (!(fabsf((float)(j - imax)) * a - T < 0.0f)) atomicAdd(crow + j, 1.0f);
    }
}

// -------------------------------------------------------------------------
extern "C" void kernel_launch(void* const* d_in, const int* in_sizes, int n_in,
                              void* d_out, int out_size, void* d_ws, size_t ws_size,
                              hipStream_t stream) {
    const float* query  = (const float*)d_in[0];
    const float* key_in = (const float*)d_in[1];
    const float* value  = (const float*)d_in[2];
    const float* W      = (const float*)d_in[3];
    const float* bias   = (const float*)d_in[4];
    const float* out_w  = (const float*)d_in[5];
    const float* out_b  = (const float*)d_in[6];

    float* out_attn = (float*)d_out;              // [B,L,E] = 4194304 f32
    float* out_cons = out_attn + ARR4M;           // [L,L]   = 1048576 f32

    _Float16* wsh = (_Float16*)d_ws;
    _Float16* qv  = wsh + 0 * (size_t)ARR4M;
    _Float16* kv  = wsh + 1 * (size_t)ARR4M;
    _Float16* vv  = wsh + 2 * (size_t)ARR4M;
    _Float16* qmv = wsh + 3 * (size_t)ARR4M;
    _Float16* kmv = wsh + 4 * (size_t)ARR4M;
    _Float16* ctx = wsh + 5 * (size_t)ARR4M;

    zero_kernel<<<dim3(1048576 / 256), 256, 0, stream>>>(out_cons, 1048576);
    proj_kernel<<<dim3(64, 16, 5), 128, 0, stream>>>(query, key_in, value, W, bias, wsh);
    attn_kernel<<<1024, 128, 0, stream>>>(qv, kv, vv, ctx);
    outproj_kernel<<<dim3(64, 16), 128, 0, stream>>>(ctx, out_w, out_b, out_attn);
    mask_kernel<<<4096, 256, 0, stream>>>(qmv, kmv, out_cons);
}
